// SpikingParrot_29970281791988
// MI455X (gfx1250) — compile-verified, matrix operand order checked by reading, not run on hardware
//
#include <hip/hip_runtime.h>
#include <cmath>

#define B_   4
#define S_   256
#define H_   512
#define V_   32000
#define L_   2
#define G_   2048
#define BS_  (B_ * S_)
#define NBLK 16   // workgroups participating in one recurrence

typedef __attribute__((ext_vector_type(16))) __bf16 v16bf;
typedef __attribute__((ext_vector_type(8)))  __bf16 v8bf;
typedef __attribute__((ext_vector_type(8)))  float  v8f;
typedef __attribute__((ext_vector_type(4)))  float  v4f;

static __device__ __forceinline__ float sigm_(float x) { return 1.0f / (1.0f + __expf(-x)); }

// ---- WMMA tile loaders (wave32 layouts per CDNA5 ISA) -------------
// 16-bit A (16x32) / B-as-Wt (rows of row-major [N,K] weight): lane l holds
// row/col (l&15); k elements {klo..klo+7, 16+klo..23+klo}, klo = 8*(l>>4).

static __device__ __forceinline__ v16bf ld_tile_bf16(const __bf16* p, int ld,
                                                     int r0, int k0, int lane) {
  const int r   = r0 + (lane & 15);
  const int klo = (lane >> 4) << 3;
  const __bf16* q = p + (size_t)r * ld + k0 + klo;
  union { v16bf v; v8bf h[2]; } u;
  u.h[0] = *(const v8bf*)(q);
  u.h[1] = *(const v8bf*)(q + 16);
  return u.v;
}

static __device__ __forceinline__ v16bf ld_tile_f32cvt(const float* p, int ld,
                                                       int r0, int k0, int lane) {
  const int r   = r0 + (lane & 15);
  const int klo = (lane >> 4) << 3;
  const float* q = p + (size_t)r * ld + k0 + klo;
  v4f f0 = *(const v4f*)(q);
  v4f f1 = *(const v4f*)(q + 4);
  v4f f2 = *(const v4f*)(q + 16);
  v4f f3 = *(const v4f*)(q + 20);
  v16bf o;
#pragma unroll
  for (int i = 0; i < 4; ++i) {
    o[i]      = (__bf16)f0[i];
    o[4 + i]  = (__bf16)f1[i];
    o[8 + i]  = (__bf16)f2[i];
    o[12 + i] = (__bf16)f3[i];
  }
  return o;
}

// ---- GEMM: C[M,N] = A[M,K](bf16) * B[N,K]^T(f32, cvt in-reg) + bias ----------
// block = 8 waves; wave w owns N-tile n0 = bx*128 + 16w, M-block of 64 reuses B.
__global__ __launch_bounds__(256)
void gemm_bf16a_f32b(const __bf16* __restrict__ A, const float* __restrict__ Bw,
                     const float* __restrict__ bias, float* __restrict__ C,
                     int M, int N, int K) {
  const int lane = threadIdx.x & 31;
  const int wave = threadIdx.x >> 5;
  const int n0 = blockIdx.x * 128 + wave * 16;
  const int m0 = blockIdx.y * 64;
  (void)M;

  v8f acc[4];
#pragma unroll
  for (int mt = 0; mt < 4; ++mt)
#pragma unroll
    for (int i = 0; i < 8; ++i) acc[mt][i] = 0.0f;

  for (int k0 = 0; k0 < K; k0 += 32) {
    v16bf bt = ld_tile_f32cvt(Bw, K, n0, k0, lane);
#pragma unroll
    for (int mt = 0; mt < 4; ++mt) {
      v16bf at = ld_tile_bf16(A, K, m0 + mt * 16, k0, lane);
      acc[mt] = __builtin_amdgcn_wmma_f32_16x16x32_bf16(
          false, at, false, bt, (short)0, acc[mt], false, false);
    }
  }

  const int nn   = n0 + (lane & 15);
  const int vrow = (lane >> 4) << 3;
  const float bv = bias ? bias[nn] : 0.0f;
#pragma unroll
  for (int mt = 0; mt < 4; ++mt)
#pragma unroll
    for (int v = 0; v < 8; ++v)
      C[(size_t)(m0 + mt * 16 + vrow + v) * N + nn] = acc[mt][v] + bv;
}

// ---- SLSTM recurrence: 16 blocks x 8 waves; Whh slice held in VGPRs ----------
// Block blk owns hidden units [32*blk, 32*blk+32); wave w owns gate tile
// rows nrow0 = (w>>1)*H + 32*blk + (w&1)*16 of the 4H gate vector.
// h (padded to M=16 rows, bf16) is double-buffered in global; one monotonic
// atomic barrier across the 16 blocks per timestep.
__global__ __launch_bounds__(256)
void slstm_rec(const float* __restrict__ xg,    // [B*S, G], row b*S+t
               const float* __restrict__ Whh,   // [G, H] f32
               const float* __restrict__ bhh,   // [G]
               const float* __restrict__ thr,   // [H] or null
               const float* __restrict__ c_init,// [B,H] or null
               __bf16* __restrict__ h_buf,      // [2][16][H] bf16 (pre-inited)
               __bf16* __restrict__ spk,        // [B,S,H] or null
               float* __restrict__ syn_out,     // [B,H] or null
               float* __restrict__ mem_out,     // [B,H] or null
               int* __restrict__ cnt, int steps) {
  const int tid  = threadIdx.x;
  const int lane = tid & 31;
  const int wave = tid >> 5;
  const int blk  = blockIdx.x;
  const int j0   = blk * 32;
  const int gate = wave >> 1;
  const int nrow0 = gate * H_ + j0 + (wave & 1) * 16;

  __shared__ float g_lds[4][B_][32];
  __shared__ float c_lds[B_][32];

  if (tid < B_ * 32) {
    const int b = tid >> 5, u = tid & 31;
    c_lds[b][u] = c_init ? c_init[b * H_ + j0 + u] : 0.0f;
  }

  // Pin the recurrent weight slice (16 gate rows x K=512, bf16) in 128 VGPRs.
  v16bf bq[H_ / 32];
#pragma unroll
  for (int kk = 0; kk < H_ / 32; ++kk)
    bq[kk] = ld_tile_f32cvt(Whh, H_, nrow0, kk * 32, lane);

  const int gr = nrow0 + (lane & 15);
  const float bhh_c = bhh[gr];
  __syncthreads();

  for (int t = 0; t < steps; ++t) {
    const __bf16* hc = h_buf + (size_t)(t & 1) * 16 * H_;
    v8f acc;
#pragma unroll
    for (int i = 0; i < 8; ++i) acc[i] = 0.0f;
#pragma unroll
    for (int kk = 0; kk < H_ / 32; ++kk) {
      v16bf at = ld_tile_bf16(hc, H_, 0, kk * 32, lane);
      acc = __builtin_amdgcn_wmma_f32_16x16x32_bf16(
          false, at, false, bq[kk], (short)0, acc, false, false);
    }
    if (lane < 16) {  // C rows 0..3 live in VGPRs 0..3 of lanes 0..15
#pragma unroll
      for (int v = 0; v < B_; ++v) {
        const float pre = acc[v] + bhh_c + xg[(size_t)(v * S_ + t) * G_ + gr];
        g_lds[gate][v][(wave & 1) * 16 + (lane & 15)] = pre;
      }
    }
    __syncthreads();

    if (tid < B_ * 32) {
      const int b = tid >> 5, u = tid & 31;
      const float iv = g_lds[0][b][u], fv = g_lds[1][b][u];
      const float gv = g_lds[2][b][u], ov = g_lds[3][b][u];
      const float cn = sigm_(fv) * c_lds[b][u] + sigm_(iv) * tanhf(gv);
      const float hn = sigm_(ov) * tanhf(cn);
      c_lds[b][u] = cn;
      h_buf[(size_t)((t + 1) & 1) * 16 * H_ + b * H_ + j0 + u] = (__bf16)hn;
      if (spk) {
        const float s = (hn - thr[j0 + u]) > 0.0f ? 1.0f : 0.0f;
        spk[(size_t)(b * S_ + t) * H_ + j0 + u] = (__bf16)s;
      }
      if (syn_out && t == steps - 1) {
        syn_out[b * H_ + j0 + u] = cn;
        mem_out[b * H_ + j0 + u] = hn;
      }
    }
    __threadfence();
    __syncthreads();
    if (tid == 0) {
      __hip_atomic_fetch_add(cnt, 1, __ATOMIC_RELEASE, __HIP_MEMORY_SCOPE_AGENT);
      const int target = NBLK * (t + 1);
      while (__hip_atomic_load(cnt, __ATOMIC_ACQUIRE, __HIP_MEMORY_SCOPE_AGENT) < target)
        __builtin_amdgcn_s_sleep(8);
    }
    __syncthreads();
  }
}

// ---- elementwise helpers -----------------------------------------------------
__global__ void embed_dyt(const int* __restrict__ toks, const float* __restrict__ emb,
                          const float* __restrict__ alpha, const float* __restrict__ gamma,
                          const float* __restrict__ beta, float* __restrict__ xf,
                          __bf16* __restrict__ xb) {
  const int idx = blockIdx.x * blockDim.x + threadIdx.x;
  if (idx >= BS_ * H_) return;
  const int row = idx / H_, j = idx % H_;
  const int tok = toks[row];
  const float e = (tok == 0) ? 0.0f : emb[(size_t)tok * H_ + j];  // padding_idx
  const float v = gamma[j] * tanhf(alpha[0] * e) + beta[j];
  if (xf) xf[idx] = v;
  xb[idx] = (__bf16)v;
}

__global__ void resid_dyt(const float* __restrict__ y, const float* __restrict__ xprev,
                          const float* __restrict__ alpha, const float* __restrict__ gamma,
                          const float* __restrict__ beta, float* __restrict__ xf,
                          __bf16* __restrict__ xb) {
  const int idx = blockIdx.x * blockDim.x + threadIdx.x;
  if (idx >= BS_ * H_) return;
  const int j = idx % H_;
  const float v0 = y[idx] + (xprev ? xprev[idx] : 0.0f);
  const float v = gamma[j] * tanhf(alpha[0] * v0) + beta[j];
  xf[idx] = v;
  xb[idx] = (__bf16)v;
}

__global__ void init_h(const float* __restrict__ h_init, __bf16* __restrict__ h_buf) {
  const int idx = blockIdx.x * blockDim.x + threadIdx.x;
  if (idx >= 2 * 16 * H_) return;
  const int buf = idx / (16 * H_);
  const int r = (idx / H_) & 15, j = idx % H_;
  float v = 0.0f;
  if (buf == 0 && r < B_ && h_init) v = h_init[r * H_ + j];
  h_buf[idx] = (__bf16)v;
}

// ---- host orchestration ------------------------------------------------------
extern "C" void kernel_launch(void* const* d_in, const int* in_sizes, int n_in,
                              void* d_out, int out_size, void* d_ws, size_t ws_size,
                              hipStream_t stream) {
  (void)in_sizes; (void)n_in; (void)out_size; (void)ws_size;
  const int*   src       = (const int*)d_in[0];
  const int*   tgt       = (const int*)d_in[1];
  const float* emb_enc   = (const float*)d_in[2];
  const float* enc_alpha = (const float*)d_in[3];
  const float* enc_gamma = (const float*)d_in[4];
  const float* enc_beta  = (const float*)d_in[5];
  const float* enc_Wih   = (const float*)d_in[6];
  const float* enc_Whh   = (const float*)d_in[7];
  const float* enc_bih   = (const float*)d_in[8];
  const float* enc_bhh   = (const float*)d_in[9];
  /* enc_thr d_in[10] unused: encoder spikes are discarded */
  const float* emb_dec   = (const float*)d_in[11];
  const float* dec_alpha = (const float*)d_in[12];
  const float* dec_gamma = (const float*)d_in[13];
  const float* dec_beta  = (const float*)d_in[14];
  const float* dec_Wih   = (const float*)d_in[15];
  const float* dec_Whh   = (const float*)d_in[16];
  const float* dec_bih   = (const float*)d_in[17];
  const float* dec_bhh   = (const float*)d_in[18];
  const float* dec_thr   = (const float*)d_in[19];
  const float* fc_W      = (const float*)d_in[20];
  const float* fc_b      = (const float*)d_in[21];
  const float* dyt_alpha = (const float*)d_in[22];
  const float* dyt_gamma = (const float*)d_in[23];
  const float* dyt_beta  = (const float*)d_in[24];
  const float* out_W     = (const float*)d_in[25];
  const float* out_b     = (const float*)d_in[26];
  float* out = (float*)d_out;

  char* p = (char*)d_ws;
  auto alloc = [&](size_t bytes) -> char* {
    char* r = p; p += (bytes + 255) & ~(size_t)255; return r;
  };
  int*    counters = (int*)   alloc(64);
  __bf16* h_buf    = (__bf16*)alloc((size_t)2 * 16 * H_ * 2);
  __bf16* xenc_bf  = (__bf16*)alloc((size_t)BS_ * H_ * 2);
  float*  x_f32    = (float*) alloc((size_t)BS_ * H_ * 4);
  __bf16* x_bf     = (__bf16*)alloc((size_t)BS_ * H_ * 2);
  float*  xg       = (float*) alloc((size_t)BS_ * G_ * 4);
  __bf16* spk      = (__bf16*)alloc((size_t)BS_ * H_ * 2);
  float*  y        = (float*) alloc((size_t)BS_ * H_ * 4);
  float*  syn      = (float*) alloc((size_t)L_ * B_ * H_ * 4);
  float*  mem      = (float*) alloc((size_t)L_ * B_ * H_ * 4);

  hipMemsetAsync(counters, 0, 64, stream);

  const int EW = (BS_ * H_ + 255) / 256;
  const dim3 gxg(G_ / 128, BS_ / 64), gfc(H_ / 128, BS_ / 64), gout(V_ / 128, BS_ / 64);

  // ---- encoder (every layer consumes src_e, per reference) ----
  embed_dyt<<<EW, 256, 0, stream>>>(src, emb_enc, enc_alpha, enc_gamma, enc_beta,
                                    nullptr, xenc_bf);
  for (int l = 0; l < L_; ++l) {
    gemm_bf16a_f32b<<<gxg, 256, 0, stream>>>(xenc_bf, enc_Wih + (size_t)l * G_ * H_,
                                             enc_bih + l * G_, xg, BS_, G_, H_);
    init_h<<<(2 * 16 * H_) / 256, 256, 0, stream>>>(nullptr, h_buf);
    slstm_rec<<<NBLK, 256, 0, stream>>>(xg, enc_Whh + (size_t)l * G_ * H_,
                                        enc_bhh + l * G_, nullptr, nullptr, h_buf,
                                        nullptr, syn + l * B_ * H_, mem + l * B_ * H_,
                                        counters + l, S_);
  }

  // ---- decoder ----
  embed_dyt<<<EW, 256, 0, stream>>>(tgt, emb_dec, dec_alpha, dec_gamma, dec_beta,
                                    x_f32, x_bf);
  for (int l = 0; l < L_; ++l) {
    gemm_bf16a_f32b<<<gxg, 256, 0, stream>>>(x_bf, dec_Wih + (size_t)l * G_ * H_,
                                             dec_bih + l * G_, xg, BS_, G_, H_);
    init_h<<<(2 * 16 * H_) / 256, 256, 0, stream>>>(mem + l * B_ * H_, h_buf);
    slstm_rec<<<NBLK, 256, 0, stream>>>(xg, dec_Whh + (size_t)l * G_ * H_,
                                        dec_bhh + l * G_, dec_thr + l * H_,
                                        syn + l * B_ * H_, h_buf, spk,
                                        nullptr, nullptr, counters + 2 + l, S_);
    gemm_bf16a_f32b<<<gfc, 256, 0, stream>>>(spk, fc_W + (size_t)l * H_ * H_,
                                             fc_b + l * H_, y, BS_, H_, H_);
    resid_dyt<<<EW, 256, 0, stream>>>(y, l > 0 ? x_f32 : nullptr, dyt_alpha + l,
                                      dyt_gamma + l * H_, dyt_beta + l * H_,
                                      x_f32, x_bf);
  }

  // ---- logits: [1024,512]x[32000,512]^T ----
  gemm_bf16a_f32b<<<gout, 256, 0, stream>>>(x_bf, out_W, out_b, out, BS_, V_, H_);
}
